// kws_bnn3_57775900066230
// MI455X (gfx1250) — 8-bit matrix operand order repaired and checked by reading, compiled, not run on hardware
//
#include <hip/hip_runtime.h>
#include <stdint.h>

typedef __attribute__((ext_vector_type(8))) int v8i;

#define THREADS 256
#define SPT 2
#define ROW_STRIDE 136

__device__ __forceinline__ void pack_input_bits(const float* __restrict__ xs,
                                                uint32_t colbits[6]) {
  const float4* xp = (const float4*)xs;
#pragma unroll
  for (int i = 0; i < 6; ++i) colbits[i] = 0u;
#pragma unroll
  for (int i = 0; i < 30; ++i) {
    float4 v = xp[i];
    const int e = 4 * i;
    colbits[(e + 0) % 6] |= (__float_as_uint(v.x) >> 31) << ((e + 0) / 6);
    colbits[(e + 1) % 6] |= (__float_as_uint(v.y) >> 31) << ((e + 1) / 6);
    colbits[(e + 2) % 6] |= (__float_as_uint(v.z) >> 31) << ((e + 2) / 6);
    colbits[(e + 3) % 6] |= (__float_as_uint(v.w) >> 31) << ((e + 3) / 6);
  }
}

__device__ __forceinline__ void load_conv_wbits(const float* __restrict__ convw,
                                                uint32_t wb[3]) {
#pragma unroll
  for (int c = 0; c < 3; ++c) {
    uint32_t b = 0;
#pragma unroll
    for (int i = 0; i < 5; ++i)
      b |= (__float_as_uint(convw[c * 5 + i]) >> 31) << i;
    wb[c] = b;
  }
}

__device__ __forceinline__ void conv_to_lds_row(const uint32_t colbits[6],
                                                const uint32_t wb[3],
                                                uint32_t* rowp) {
  uint32_t word = 0;
  int k = 0;
#pragma unroll
  for (int c = 0; c < 3; ++c)
#pragma unroll
    for (int oh = 0; oh < 6; ++oh)
#pragma unroll
      for (int ow = 0; ow < 6; ++ow) {
        uint32_t win = (colbits[ow] >> (3 * oh)) & 31u;
        int pc = __popc(win ^ wb[c]);
        uint32_t byte = (pc <= 2) ? 0x01u : 0xFFu;
        word |= byte << (8 * (k & 3));
        if ((k & 3) == 3) {
          rowp[k >> 2] = word;
          word = 0;
        }
        ++k;
      }
#pragma unroll
  for (int wdx = 27; wdx < 32; ++wdx) rowp[wdx] = 0u;
}

__global__ __launch_bounds__(THREADS) void bnn_wmma_kernel(
    const float* __restrict__ x, const float* __restrict__ convw,
    const float* __restrict__ fcw, const float* __restrict__ fcb,
    float* __restrict__ out) {
  __shared__ uint8_t sA[SPT * THREADS * ROW_STRIDE];
  __shared__ int slog[SPT * THREADS][2];

  const int tid = threadIdx.x;
  const int lane = tid & 31;
  const int wv = tid >> 5;
  const int n = lane & 15;
  const int kh = lane >> 4;
  const size_t base = (size_t)blockIdx.x * (THREADS * SPT);

  uint32_t wb[3];
  load_conv_wbits(convw, wb);

#pragma unroll
  for (int s = 0; s < SPT; ++s) {
    const size_t sample = base + s * THREADS + tid;
    uint32_t colbits[6];
    pack_input_bits(x + sample * 120, colbits);
    conv_to_lds_row(colbits, wb,
                    (uint32_t*)(sA + (s * THREADS + tid) * ROW_STRIDE));
  }

  v8i bf0, bf1;
#pragma unroll
  for (int v = 0; v < 8; ++v) {
    uint32_t w0 = 0, w1 = 0;
#pragma unroll
    for (int j = 0; j < 4; ++j) {
      int k0 = (v >> 1) * 16 + kh * 8 + (v & 1) * 4 + j;
      int k1 = k0 + 64;
      uint32_t b0 = 0, b1 = 0;
      if (n < 2) {
        b0 = (__float_as_uint(fcw[n * 108 + k0]) >> 31) ? 0xFFu : 0x01u;
        if (k1 < 108)
          b1 = (__float_as_uint(fcw[n * 108 + k1]) >> 31) ? 0xFFu : 0x01u;
      }
      w0 |= b0 << (8 * j);
      w1 |= b1 << (8 * j);
    }
    bf0[v] = (int)w0;
    bf1[v] = (int)w1;
  }

  __syncthreads();

#pragma unroll
  for (int s = 0; s < SPT; ++s) {
#pragma unroll
    for (int t = 0; t < 2; ++t) {
      const int row = s * THREADS + (wv << 5) + (t << 4) + n;
      const unsigned long long* p =
          (const unsigned long long*)(sA + row * ROW_STRIDE + kh * 8);
      unsigned long long q0 = p[0], q1 = p[2], q2 = p[4], q3 = p[6];
      unsigned long long r0 = p[8], r1 = p[10], r2 = p[12], r3 = p[14];
      v8i a0 = {(int)q0, (int)(q0 >> 32), (int)q1, (int)(q1 >> 32),
                (int)q2, (int)(q2 >> 32), (int)q3, (int)(q3 >> 32)};
      v8i a1 = {(int)r0, (int)(r0 >> 32), (int)r1, (int)(r1 >> 32),
                (int)r2, (int)(r2 >> 32), (int)r3, (int)(r3 >> 32)};
      v8i acc = {0, 0, 0, 0, 0, 0, 0, 0};
      acc = __builtin_amdgcn_wmma_i32_16x16x64_iu8(true, a0, true, bf0, acc,
                                                   false, false);
      acc = __builtin_amdgcn_wmma_i32_16x16x64_iu8(true, a1, true, bf1, acc,
                                                   false, false);
      if (n < 2) {
        const int sbase = s * THREADS + (wv << 5) + (t << 4) + (kh << 3);
#pragma unroll
        for (int r = 0; r < 8; ++r) slog[sbase + r][n] = acc[r];
      }
    }
  }

  __syncthreads();

  const float b0f = fcb[0], b1f = fcb[1];
#pragma unroll
  for (int s = 0; s < SPT; ++s) {
    const int slot = s * THREADS + tid;
    const size_t sample = base + slot;
    const float l0 = (float)slog[slot][0] + b0f;
    const float l1 = (float)slog[slot][1] + b1f;
    const float m = fmaxf(l0, l1);
    const float lse = m + __logf(__expf(l0 - m) + __expf(l1 - m));
    float2 o = make_float2(l0 - lse, l1 - lse);
    *(float2*)(out + sample * 2) = o;
  }
}

__global__ __launch_bounds__(THREADS) void bnn_tail_kernel(
    const float* __restrict__ x, const float* __restrict__ convw,
    const float* __restrict__ fcw, const float* __restrict__ fcb,
    float* __restrict__ out, int start, int total) {
  const int s = start + blockIdx.x * THREADS + threadIdx.x;
  if (s >= total) return;

  uint32_t wb[3];
  load_conv_wbits(convw, wb);
  uint32_t colbits[6];
  pack_input_bits(x + (size_t)s * 120, colbits);

  uint32_t abits[4] = {0, 0, 0, 0};
  {
    int k = 0;
#pragma unroll
    for (int c = 0; c < 3; ++c)
#pragma unroll
      for (int oh = 0; oh < 6; ++oh)
#pragma unroll
        for (int ow = 0; ow < 6; ++ow) {
          uint32_t win = (colbits[ow] >> (3 * oh)) & 31u;
          uint32_t neg = (__popc(win ^ wb[c]) >= 3) ? 1u : 0u;
          abits[k >> 5] |= neg << (k & 31);
          ++k;
        }
  }

  float lg[2];
#pragma unroll
  for (int j = 0; j < 2; ++j) {
    uint32_t wbt[4] = {0, 0, 0, 0};
    for (int k = 0; k < 108; ++k)
      wbt[k >> 5] |= (__float_as_uint(fcw[j * 108 + k]) >> 31) << (k & 31);
    int diff = __popc(abits[0] ^ wbt[0]) + __popc(abits[1] ^ wbt[1]) +
               __popc(abits[2] ^ wbt[2]) + __popc(abits[3] ^ wbt[3]);
    lg[j] = (float)(108 - 2 * diff) + fcb[j];
  }
  const float m = fmaxf(lg[0], lg[1]);
  const float lse = m + __logf(__expf(lg[0] - m) + __expf(lg[1] - m));
  out[(size_t)s * 2 + 0] = lg[0] - lse;
  out[(size_t)s * 2 + 1] = lg[1] - lse;
}

extern "C" void kernel_launch(void* const* d_in, const int* in_sizes, int n_in,
                              void* d_out, int out_size, void* d_ws,
                              size_t ws_size, hipStream_t stream) {
  (void)n_in;
  (void)out_size;
  (void)d_ws;
  (void)ws_size;
  const float* x = (const float*)d_in[0];
  const float* convw = (const float*)d_in[1];
  const float* fcw = (const float*)d_in[2];
  const float* fcb = (const float*)d_in[3];
  float* out = (float*)d_out;

  const long long B = (long long)in_sizes[0] / 120;
  const long long per_block = THREADS * SPT;
  const long long full = B / per_block;
  if (full > 0) {
    bnn_wmma_kernel<<<dim3((unsigned)full), dim3(THREADS), 0, stream>>>(
        x, convw, fcw, fcb, out);
  }
  const long long rem = B - full * per_block;
  if (rem > 0) {
    bnn_tail_kernel<<<dim3((unsigned)((rem + THREADS - 1) / THREADS)),
                      dim3(THREADS), 0, stream>>>(
        x, convw, fcw, fcb, out, (int)(full * per_block), (int)B);
  }
}
